// QLoRA_47029891891910
// MI455X (gfx1250) — matrix operand order repaired and checked by reading, not compiled, not run on hardware
//
#include <hip/hip_runtime.h>

#define IN_F  4096
#define OUT_F 8192
#define SEQ   512
#define RANK  8

typedef __attribute__((ext_vector_type(16))) _Float16 v16h;
typedef __attribute__((ext_vector_type(8)))  _Float16 v8h;
typedef __attribute__((ext_vector_type(4)))  _Float16 v4h;
typedef __attribute__((ext_vector_type(8)))  float    v8f;

// LDS B-tile: 64 columns x 32 K-halves, padded to 40 halves (80 B) per column:
// stride = 20 banks -> the 16-lane b128 fan-out hits distinct bank groups.
#define BS_STRIDE 40

// ---------------------------------------------------------------------------
// Kernel 1: convert x (f32) -> xh (f16) once.  512*4096 elems, 4 per thread.
// ---------------------------------------------------------------------------
__global__ void __launch_bounds__(256)
xcvt_kernel(const float* __restrict__ x, _Float16* __restrict__ xh)
{
    int i = (blockIdx.x * 256 + threadIdx.x) * 4;
    float4 v = *(const float4*)(x + i);
    v4h h;
    h[0] = (_Float16)v.x;
    h[1] = (_Float16)v.y;
    h[2] = (_Float16)v.z;
    h[3] = (_Float16)v.w;
    *(v4h*)(xh + i) = h;
}

// ---------------------------------------------------------------------------
// Kernel 2: t[512][8] = x @ A   (one block per sequence row, LDS reduction)
// ---------------------------------------------------------------------------
__global__ void __launch_bounds__(256)
xa_kernel(const float* __restrict__ x, const float* __restrict__ A,
          float* __restrict__ t)
{
    __shared__ float red[RANK][256];
    const int s   = blockIdx.x;
    const int tid = threadIdx.x;

    float acc[RANK] = {};
    for (int i = tid; i < IN_F; i += 256) {
        float xv = x[(size_t)s * IN_F + i];
        const float4* ap = (const float4*)(A + (size_t)i * RANK);
        float4 a0 = ap[0], a1 = ap[1];
        acc[0] += xv * a0.x;  acc[1] += xv * a0.y;
        acc[2] += xv * a0.z;  acc[3] += xv * a0.w;
        acc[4] += xv * a1.x;  acc[5] += xv * a1.y;
        acc[6] += xv * a1.z;  acc[7] += xv * a1.w;
    }
    #pragma unroll
    for (int r = 0; r < RANK; ++r) red[r][tid] = acc[r];
    __syncthreads();
    for (int off = 128; off > 0; off >>= 1) {
        if (tid < off) {
            #pragma unroll
            for (int r = 0; r < RANK; ++r) red[r][tid] += red[r][tid + off];
        }
        __syncthreads();
    }
    if (tid < RANK) t[(size_t)s * RANK + tid] = red[tid][0];
}

// ---------------------------------------------------------------------------
// Kernel 3: out = t @ B   (LoRA term written first; GEMM then accumulates)
// ---------------------------------------------------------------------------
__global__ void __launch_bounds__(256)
lora_kernel(const float* __restrict__ t, const float* __restrict__ B,
            float* __restrict__ out)
{
    int idx = blockIdx.x * 256 + threadIdx.x;        // over 512*8192
    int s = idx >> 13;
    int o = idx & (OUT_F - 1);
    const float* ts = t + (size_t)s * RANK;
    float acc = 0.f;
    #pragma unroll
    for (int r = 0; r < RANK; ++r) acc += ts[r] * B[(size_t)r * OUT_F + o];
    out[idx] = acc;
}

// ---------------------------------------------------------------------------
// Stage one 64x32 dequantized B tile into LDS (whole 256-thread block).
// Each thread: 8 consecutive int32 codes -> 8 f16 -> one ds_store_b128.
// ---------------------------------------------------------------------------
__device__ __forceinline__ void stage_btile(const int* __restrict__ qwt,
                                            _Float16* __restrict__ dst,
                                            int tid, int k0, float sc, float zs)
{
    const int c  = tid >> 2;                 // column 0..63
    const int ko = (tid & 3) * 8;            // k offset 0,8,16,24
    const int* wp = qwt + (size_t)c * IN_F + k0 + ko;
    int4 q0 = ((const int4*)wp)[0];
    int4 q1 = ((const int4*)wp)[1];
    v8h h;
    h[0] = (_Float16)((float)q0.x * sc - zs);
    h[1] = (_Float16)((float)q0.y * sc - zs);
    h[2] = (_Float16)((float)q0.z * sc - zs);
    h[3] = (_Float16)((float)q0.w * sc - zs);
    h[4] = (_Float16)((float)q1.x * sc - zs);
    h[5] = (_Float16)((float)q1.y * sc - zs);
    h[6] = (_Float16)((float)q1.z * sc - zs);
    h[7] = (_Float16)((float)q1.w * sc - zs);
    *(v8h*)(dst + c * BS_STRIDE + ko) = h;
}

// ---------------------------------------------------------------------------
// One K=32 compute step: 2 A-frags from global, 4 B-frags from LDS, 8 WMMAs.
// ---------------------------------------------------------------------------
__device__ __forceinline__ void compute_step(const _Float16* __restrict__ bufc,
                                             const _Float16* __restrict__ a0p,
                                             const _Float16* __restrict__ a1p,
                                             int k0, int lm, int half,
                                             v8f acc[2][4])
{
    v16h afr[2];
    {
        v8h lo = *(const v8h*)(a0p + k0);
        v8h hi = *(const v8h*)(a0p + k0 + 16);
        #pragma unroll
        for (int q = 0; q < 8; ++q) { afr[0][q] = lo[q]; afr[0][8 + q] = hi[q]; }
    }
    {
        v8h lo = *(const v8h*)(a1p + k0);
        v8h hi = *(const v8h*)(a1p + k0 + 16);
        #pragma unroll
        for (int q = 0; q < 8; ++q) { afr[1][q] = lo[q]; afr[1][8 + q] = hi[q]; }
    }

    #pragma unroll
    for (int j = 0; j < 4; ++j) {
        const _Float16* bsrc = bufc + (16 * j + lm) * BS_STRIDE + 8 * half;
        v8h lo = *(const v8h*)bsrc;
        v8h hi = *(const v8h*)(bsrc + 16);
        v16h bfr;
        #pragma unroll
        for (int q = 0; q < 8; ++q) { bfr[q] = lo[q]; bfr[8 + q] = hi[q]; }

        acc[0][j] = __builtin_amdgcn_wmma_f32_16x16x32_f16(
            false, afr[0], false, bfr, (short)0, acc[0][j], false, false);
        acc[1][j] = __builtin_amdgcn_wmma_f32_16x16x32_f16(
            false, afr[1], false, bfr, (short)0, acc[1][j], false, false);
    }
}

// ---------------------------------------------------------------------------
// Kernel 4: out += xh @ dequant(Wq)^T  via v_wmma_f32_16x16x32_f16.
// Block (8 waves) owns a 256(M) x 64(N) output tile; dequantized B tiles are
// staged in double-buffered LDS shared by all 8 waves.  The K-loop is
// manually unrolled 2x with statically-named buffers and NO internal
// branches (final stage address is clamped), so accumulators stay pinned.
// ---------------------------------------------------------------------------
__global__ void __launch_bounds__(256)
qgemm_kernel(const _Float16* __restrict__ xh,
             const int*      __restrict__ qw,
             const float*    __restrict__ scale_p,
             const int*      __restrict__ zp_p,
             float*          __restrict__ out)
{
    __shared__ _Float16 Bs0[64 * BS_STRIDE];     // 5 KB
    __shared__ _Float16 Bs1[64 * BS_STRIDE];     // 5 KB

    const float sc = *scale_p;
    const float zs = sc * (float)(*zp_p);        // dequant(q) = q*sc - zs

    const int tid  = threadIdx.x;
    const int lane = tid & 31;
    const int wave = tid >> 5;
    const int half = lane >> 4;                  // lane group 0..15 / 16..31
    const int lm   = lane & 15;

    const int tileN = blockIdx.x * 64;
    const int tileM = blockIdx.y * 256 + wave * 32;

    const int* qwt = qw + (size_t)tileN * IN_F;  // block's weight rows

    v8f acc[2][4] = {};                          // 8 accumulators, f32

    // A-operand rows (ISA 16-bit A 16x32 layout): m = lane%16,
    // halves 0..7 at K = k0 + 8*half, halves 8..15 at K + 16.
    const _Float16* a0p = xh + (size_t)(tileM + lm) * IN_F + 8 * half;
    const _Float16* a1p = a0p + (size_t)16 * IN_F;

    stage_btile(qwt, Bs0, tid, 0, sc, zs);
    __syncthreads();

    #pragma unroll 1
    for (int k0 = 0; k0 < IN_F; k0 += 64) {
        // ---- half A: stage slice k0+32 into Bs1, compute slice k0 from Bs0
        stage_btile(qwt, Bs1, tid, k0 + 32, sc, zs);
        compute_step(Bs0, a0p, a1p, k0, lm, half, acc);
        __syncthreads();

        // ---- half B: stage slice k0+64 (clamped; last restage is unused)
        const int kn = (k0 + 64 < IN_F) ? (k0 + 64) : (IN_F - 32);
        stage_btile(qwt, Bs0, tid, kn, sc, zs);
        compute_step(Bs1, a0p, a1p, k0 + 32, lm, half, acc);
        __syncthreads();
    }

    // ---- epilogue: out(+lora already there) += acc ----
    // C/D layout: vgpr r -> row m = r + 8*half (within 16x16 frag), col n = lane%16
    #pragma unroll
    for (int i = 0; i < 2; ++i) {
        #pragma unroll
        for (int j = 0; j < 4; ++j) {
            const int o = tileN + 16 * j + lm;
            #pragma unroll
            for (int r = 0; r < 8; ++r) {
                const int s = tileM + 16 * i + r + 8 * half;
                float* op = out + (size_t)s * OUT_F + o;
                *op = *op + acc[i][j][r];
            }
        }
    }
}

// ---------------------------------------------------------------------------
extern "C" void kernel_launch(void* const* d_in, const int* in_sizes, int n_in,
                              void* d_out, int out_size, void* d_ws, size_t ws_size,
                              hipStream_t stream)
{
    (void)in_sizes; (void)n_in; (void)out_size; (void)ws_size;

    const float* x     = (const float*)d_in[0];   // [1,512,4096] f32
    const int*   qw    = (const int*)  d_in[1];   // [8192,4096] int32 codes
    const float* scale = (const float*)d_in[2];   // scalar f32
    const int*   zp    = (const int*)  d_in[3];   // scalar int
    const float* A     = (const float*)d_in[4];   // [4096,8] f32
    const float* B     = (const float*)d_in[5];   // [8,8192] f32
    float*       out   = (float*)d_out;           // [1,512,8192] f32

    _Float16* xh = (_Float16*)d_ws;                                   // 4 MB
    float*    t  = (float*)((char*)d_ws + (size_t)SEQ * IN_F * 2);    // 16 KB

    // 1) x -> f16
    xcvt_kernel<<<(SEQ * IN_F) / (256 * 4), 256, 0, stream>>>(x, xh);
    // 2) t = x @ A
    xa_kernel<<<SEQ, 256, 0, stream>>>(x, A, t);
    // 3) out = t @ B (LoRA term)
    lora_kernel<<<(SEQ * OUT_F) / 256, 256, 0, stream>>>(t, B, out);
    // 4) out += x @ dequant(Wq)^T  (WMMA main GEMM, LDS-shared dequant)
    dim3 grid(OUT_F / 64, SEQ / 256);
    qgemm_kernel<<<grid, 256, 0, stream>>>(xh, qw, scale, zp, out);
}
